// KnnKernel_71975061946735
// MI455X (gfx1250) — matrix operand order repaired and checked by reading, not compiled, not run on hardware
//
#include <hip/hip_runtime.h>
#include <hip/hip_bf16.h>

// ---------------------------------------------------------------------------
// MI455X (gfx1250) fused edge-message-passing block.
// All GEMM work runs on v_wmma_f32_16x16x32_f16 (f16 in, f32 accumulate).
// Weights are pre-swizzled into fragment-major f16 so each B fragment is one
// contiguous 32B v16h load per lane. Each load batch feeds TWO independent
// WMMA accumulator chains (two output tiles share one A-fragment set) so the
// XDL pipe gets 2x work per load stall, and edge_kernel LDS is squeezed to
// ~42KB (sF/sHid union) so ~7 single-wave blocks co-reside per WGP to hide
// the rest of the latency.
// ---------------------------------------------------------------------------

typedef __attribute__((ext_vector_type(16))) _Float16 v16h;
typedef __attribute__((ext_vector_type(8)))  float    v8f;

#define WIDTH 256
#define WACT  512
#define NHEAD 8
#define CPH   32      // WIDTH / NHEAD
#define APH   64      // WACT / NHEAD
#define GN_EPS 1e-5f

#define WMMA16(A, B, C) \
  __builtin_amdgcn_wmma_f32_16x16x32_f16(false, (A), false, (B), (short)0, (C), false, false)

// ---- WMMA fragment helpers (per CDNA5 ISA §7.12.2 layouts) -----

// A fragment (16x32 f16) from LDS, row-major with stride ld (ld even).
// lane 0-15: M=lane, K = kb..kb+7 and kb+16..kb+23 with kb=0
// lane 16-31: M=lane-16, kb=8
__device__ __forceinline__ v16h ldsA(const _Float16* base, int ld, int kofs) {
  int lane = threadIdx.x & 31;
  int m = lane & 15;
  int kb = kofs + ((lane >> 4) << 3);
  const _Float16* row = base + m * ld;
  v16h a;
#pragma unroll
  for (int p = 0; p < 8; ++p) {
    int k = kb + ((p & 3) << 1) + ((p & 4) << 2); // p<4: kb+2p ; p>=4: kb+16+2(p-4)
    a[2 * p]     = row[k];
    a[2 * p + 1] = row[k + 1];
  }
  return a;
}

// C (f32 16x16) store into LDS float buffer with stride ld.
__device__ __forceinline__ void stC_f32(float* base, int ld, int nofs, v8f acc) {
  int lane = threadIdx.x & 31;
  int n  = nofs + (lane & 15);
  int rb = (lane >> 4) * 8;
#pragma unroll
  for (int r = 0; r < 8; ++r) base[(rb + r) * ld + n] = acc[r];
}

// ---- weight swizzler: w[Nout][K] row-major -> fragment-major f16 ----------
// dst chunk index = (kt*ntiles + nt)*32 + lane ; element i within chunk.
// k = kt*32 + (i>>3)*16 + (lane>>4)*8 + (i&7) ; n = nt*16 + (lane&15)   (B is K x Nout = w^T)
__global__ void swizzle_w(const float* __restrict__ w, _Float16* __restrict__ dst,
                          int K, int Nout, int srcHeadStride, int dstHeadStride) {
  int h = blockIdx.y;
  w   += (size_t)h * srcHeadStride;
  dst += (size_t)h * dstHeadStride;
  int ntiles = Nout >> 4;
  int total  = K * Nout;
  for (int idx = blockIdx.x * blockDim.x + threadIdx.x; idx < total;
       idx += gridDim.x * blockDim.x) {
    int i    = idx & 15;
    int lane = (idx >> 4) & 31;
    int tile = idx >> 9;
    int kt = tile / ntiles, nt = tile - kt * ntiles;
    int n = nt * 16 + (lane & 15);
    int k = kt * 32 + ((i & 8) << 1) + ((lane >> 4) << 3) + (i & 7);
    dst[idx] = (_Float16)w[(size_t)n * K + k];
  }
}

__global__ void zero_f32(float* __restrict__ p, long long n) {
  long long i = (long long)blockIdx.x * blockDim.x + threadIdx.x;
  long long s = (long long)gridDim.x * blockDim.x;
  for (; i < n; i += s) p[i] = 0.0f;
}

// ---- node projection: Y = X @ W^T via WMMA --------------------------------
__global__ __launch_bounds__(128) void gemm_node(const float* __restrict__ X,
                                                 const _Float16* __restrict__ Wsw,
                                                 float* __restrict__ Y, int Ntot) {
  __shared__ _Float16 sA[16][264];
  int m0 = blockIdx.x * 16;
  for (int idx = threadIdx.x; idx < 16 * 256; idx += blockDim.x) {
    int r = idx >> 8, c = idx & 255;
    int rr = m0 + r; rr = rr < Ntot ? rr : Ntot - 1;
    sA[r][c] = (_Float16)X[(size_t)rr * 256 + c];
  }
  __syncthreads();
  int wave = threadIdx.x >> 5;
  int lane = threadIdx.x & 31;
  const v16h* B = (const v16h*)Wsw;          // K=256 -> 8 kt, ntiles=16

  v16h aA[8];
#pragma unroll
  for (int kt = 0; kt < 8; ++kt) aA[kt] = ldsA(&sA[0][0], 264, kt * 32);

  bool full = (m0 + 16 <= Ntot);
#pragma unroll 1
  for (int i = 0; i < 2; ++i) {
    int nt0 = wave + i * 8;          // this wave's tiles: wave, wave+4, +8, +12
    int nt1 = nt0 + 4;
    v8f acc0 = {}, acc1 = {};
#pragma unroll
    for (int qq = 0; qq < 2; ++qq) {
      v16h b0[4], b1[4];
#pragma unroll
      for (int q = 0; q < 4; ++q) {
        int kt = qq * 4 + q;
        b0[q] = B[(kt * 16 + nt0) * 32 + lane];
        b1[q] = B[(kt * 16 + nt1) * 32 + lane];
      }
#pragma unroll
      for (int q = 0; q < 4; ++q) {
        int kt = qq * 4 + q;
        acc0 = WMMA16(aA[kt], b0[q], acc0);
        acc1 = WMMA16(aA[kt], b1[q], acc1);
      }
    }
    int rb = (lane >> 4) * 8;
    int c0 = nt0 * 16 + (lane & 15);
    int c1 = nt1 * 16 + (lane & 15);
    if (full) {
#pragma unroll
      for (int r = 0; r < 8; ++r) {
        size_t rowoff = (size_t)(m0 + rb + r) * 256;
        Y[rowoff + c0] = acc0[r];
        Y[rowoff + c1] = acc1[r];
      }
    } else {
#pragma unroll
      for (int r = 0; r < 8; ++r) {
        int row = m0 + rb + r;
        if (row < Ntot) {
          Y[(size_t)row * 256 + c0] = acc0[r];
          Y[(size_t)row * 256 + c1] = acc1[r];
        }
      }
    }
  }
}

// ---- degree ---------------------------------------------------------------
__global__ void deg_kernel(const int* __restrict__ ei, float* __restrict__ deg,
                           int Etot) {
  int e = blockIdx.x * blockDim.x + threadIdx.x;
  if (e >= Etot) return;
  int k0 = ei[e], k1 = ei[Etot + e];
  if (k0 != k1) atomicAdd(&deg[k1], 1.0f);
}

// ---- fused per-edge pipeline: one wave handles 16 edges -------------------
// Single-wave workgroup: barriers degrade to waitcnts (no barrier resource).
// sF (f32 GroupNorm staging) and sHid (f16 hidden) are never live at the same
// time -> union them to cut LDS to ~42KB and raise blocks/WGP.
__global__ __launch_bounds__(32) void edge_kernel(
    const float* __restrict__ z_rw, const int* __restrict__ ei,
    const float* __restrict__ x0, const float* __restrict__ x1,
    const _Float16* __restrict__ swPre,
    const _Float16* __restrict__ swGateP, const _Float16* __restrict__ swValP,
    const _Float16* __restrict__ swPostP,
    const _Float16* __restrict__ swGateF, const _Float16* __restrict__ swValF,
    const _Float16* __restrict__ swPostF,
    float* __restrict__ xx, int Etot) {
  __shared__ _Float16 sIn[16][264];    // z_edge / xn (f16 A operand)
  __shared__ _Float16 sGate[16][264];  // xn + bias (ffn gate A operand)
  __shared__ _Float16 sBias[16][264];  // per-edge positional bias
  __shared__ __align__(16) char uSM[16 * 520 * 2];  // sHid f16 [16][520] == sF f32 [16][260]
  _Float16 (*sHid)[520] = (_Float16(*)[520])uSM;
  float    (*sF)[260]   = (float(*)[260])uSM;
  __shared__ int   sK0[16], sK1[16];
  __shared__ float sVal[16];

  int lane = threadIdx.x;
  int e0   = blockIdx.x * 16;
  __builtin_prefetch(swPostP, 0, 1);   // warm L2 path for the fat weights

  if (lane < 16) {
    int e = e0 + lane;
    int k0 = 0, k1 = 0; float v = 0.0f;
    if (e < Etot) { k0 = ei[e]; k1 = ei[Etot + e]; v = (k0 != k1) ? 1.0f : 0.0f; }
    sK0[lane] = k0; sK1[lane] = k1; sVal[lane] = v;
  }
  __syncthreads();

  // ---- stage 1: z_edge = [z_rw[k0] | z_rw[k1]]  -> sIn[:, 0:32] (f16) ----
#pragma unroll
  for (int it = 0; it < 4; ++it) {
    int t = it * 32 + lane;
    int row = t >> 3, src = (t >> 2) & 1, q = t & 3;
    int node = src ? sK1[row] : sK0[row];
    const float4* p = (const float4*)(z_rw + (size_t)node * 16);
    float4 u = p[q];
    int c = src * 16 + q * 4;
    sIn[row][c + 0] = (_Float16)u.x;
    sIn[row][c + 1] = (_Float16)u.y;
    sIn[row][c + 2] = (_Float16)u.z;
    sIn[row][c + 3] = (_Float16)u.w;
  }
  __syncthreads();

  // ---- stage 2: xn = z_edge @ w_pre^T  (K=32, Nout=256) -> sF (f32) ------
  {
    v16h a = ldsA(&sIn[0][0], 264, 0);
    const v16h* B = (const v16h*)swPre;  // ntiles=16, single kt
#pragma unroll 1
    for (int nt = 0; nt < 16; nt += 4) {
      v16h b0 = B[(nt + 0) * 32 + lane];
      v16h b1 = B[(nt + 1) * 32 + lane];
      v16h b2 = B[(nt + 2) * 32 + lane];
      v16h b3 = B[(nt + 3) * 32 + lane];
      v8f c0 = {}, c1 = {}, c2 = {}, c3 = {};
      c0 = WMMA16(a, b0, c0);
      c1 = WMMA16(a, b1, c1);
      c2 = WMMA16(a, b2, c2);
      c3 = WMMA16(a, b3, c3);
      stC_f32(&sF[0][0], 260, (nt + 0) * 16, c0);
      stC_f32(&sF[0][0], 260, (nt + 1) * 16, c1);
      stC_f32(&sF[0][0], 260, (nt + 2) * 16, c2);
      stC_f32(&sF[0][0], 260, (nt + 3) * 16, c3);
    }
  }
  __syncthreads();

  // ---- stage 3: GroupNorm(sF) -> sIn (f16) -------------------------------
#pragma unroll 1
  for (int t = lane; t < 128; t += 32) {
    int row = t >> 3, g = (t & 7) * 32;
    float mu = 0.0f;
#pragma unroll
    for (int c = 0; c < 32; ++c) mu += sF[row][g + c];
    mu *= (1.0f / 32.0f);
    float var = 0.0f;
#pragma unroll
    for (int c = 0; c < 32; ++c) { float d = sF[row][g + c] - mu; var += d * d; }
    float rs = rsqrtf(var * (1.0f / 32.0f) + GN_EPS);
#pragma unroll
    for (int c = 0; c < 32; ++c)
      sIn[row][g + c] = (_Float16)((sF[row][g + c] - mu) * rs);
  }
  __syncthreads();

  // ---- stage 4: hidden = relu(grouped gate) * grouped value -> sHid ------
  {
    const v16h* BG = (const v16h*)swGateP;   // per head: 4 nt chunks * 32
    const v16h* BV = (const v16h*)swValP;
#pragma unroll 1
    for (int h = 0; h < 8; ++h) {
      v16h a = ldsA(&sIn[0][0], 264, h * 32);
      v16h bg[4], bv[4];
#pragma unroll
      for (int nt = 0; nt < 4; ++nt) {
        bg[nt] = BG[h * 128 + nt * 32 + lane];
        bv[nt] = BV[h * 128 + nt * 32 + lane];
      }
#pragma unroll
      for (int nt = 0; nt < 4; ++nt) {
        v8f g = {}, v = {};
        g = WMMA16(a, bg[nt], g);
        v = WMMA16(a, bv[nt], v);
        int n  = (lane & 15), rb = (lane >> 4) * 8;
#pragma unroll
        for (int r = 0; r < 8; ++r) {
          float gr = g[r] > 0.0f ? g[r] : 0.0f;
          sHid[rb + r][h * 64 + nt * 16 + n] = (_Float16)(gr * v[r]);
        }
      }
    }
  }
  __syncthreads();

  // ---- stage 5: bias = hidden @ w_post_perw^T (K=512) -> sBias (f16) -----
  {
    v16h aH[16];
#pragma unroll
    for (int kt = 0; kt < 16; ++kt) aH[kt] = ldsA(&sHid[0][0], 520, kt * 32);
    const v16h* B = (const v16h*)swPostP;    // 16 kt, ntiles=16
#pragma unroll 1
    for (int nt = 0; nt < 16; nt += 2) {
      v8f acc0 = {}, acc1 = {};
#pragma unroll
      for (int qq = 0; qq < 4; ++qq) {
        v16h b0[4], b1[4];
#pragma unroll
        for (int q = 0; q < 4; ++q) {
          int kt = qq * 4 + q;
          b0[q] = B[(kt * 16 + nt) * 32 + lane];
          b1[q] = B[(kt * 16 + nt + 1) * 32 + lane];
        }
#pragma unroll
        for (int q = 0; q < 4; ++q) {
          int kt = qq * 4 + q;
          acc0 = WMMA16(aH[kt], b0[q], acc0);
          acc1 = WMMA16(aH[kt], b1[q], acc1);
        }
      }
      int rb = (lane >> 4) * 8;
      int n0 = nt * 16 + (lane & 15);
#pragma unroll
      for (int r = 0; r < 8; ++r) {
        sBias[rb + r][n0]      = (_Float16)acc0[r];
        sBias[rb + r][n0 + 16] = (_Float16)acc1[r];
      }
    }
  }
  __syncthreads();

  // ---- stage 6: msg = x0[k0] + x1[k1] -> sF (f32), float4 gathers --------
#pragma unroll 1
  for (int row = 0; row < 16; ++row) {
    const float4* p0 = (const float4*)(x0 + (size_t)sK0[row] * 256);
    const float4* p1 = (const float4*)(x1 + (size_t)sK1[row] * 256);
#pragma unroll
    for (int j = 0; j < 2; ++j) {
      int c4 = j * 32 + lane;
      float4 u = p0[c4], w = p1[c4];
      int c = c4 * 4;
      sF[row][c + 0] = u.x + w.x;
      sF[row][c + 1] = u.y + w.y;
      sF[row][c + 2] = u.z + w.z;
      sF[row][c + 3] = u.w + w.w;
    }
  }
  __syncthreads();

  // ---- stage 7: GroupNorm(msg); sIn = xn2, sGate = xn2 + bias ------------
#pragma unroll 1
  for (int t = lane; t < 128; t += 32) {
    int row = t >> 3, g = (t & 7) * 32;
    float mu = 0.0f;
#pragma unroll
    for (int c = 0; c < 32; ++c) mu += sF[row][g + c];
    mu *= (1.0f / 32.0f);
    float var = 0.0f;
#pragma unroll
    for (int c = 0; c < 32; ++c) { float d = sF[row][g + c] - mu; var += d * d; }
    float rs = rsqrtf(var * (1.0f / 32.0f) + GN_EPS);
#pragma unroll
    for (int c = 0; c < 32; ++c) {
      float v = (sF[row][g + c] - mu) * rs;
      sIn[row][g + c]   = (_Float16)v;
      sGate[row][g + c] = (_Float16)(v + (float)sBias[row][g + c]);
    }
  }
  __syncthreads();

  // ---- stage 8: ffn gate/value -> sHid -----------------------------------
  {
    const v16h* BG = (const v16h*)swGateF;
    const v16h* BV = (const v16h*)swValF;
#pragma unroll 1
    for (int h = 0; h < 8; ++h) {
      v16h ag = ldsA(&sGate[0][0], 264, h * 32);
      v16h av = ldsA(&sIn[0][0],   264, h * 32);
      v16h bg[4], bv[4];
#pragma unroll
      for (int nt = 0; nt < 4; ++nt) {
        bg[nt] = BG[h * 128 + nt * 32 + lane];
        bv[nt] = BV[h * 128 + nt * 32 + lane];
      }
#pragma unroll
      for (int nt = 0; nt < 4; ++nt) {
        v8f g = {}, v = {};
        g = WMMA16(ag, bg[nt], g);
        v = WMMA16(av, bv[nt], v);
        int n  = (lane & 15), rb = (lane >> 4) * 8;
#pragma unroll
        for (int r = 0; r < 8; ++r) {
          float gr = g[r] > 0.0f ? g[r] : 0.0f;
          sHid[rb + r][h * 64 + nt * 16 + n] = (_Float16)(gr * v[r]);
        }
      }
    }
  }
  __syncthreads();

  // ---- stage 9: out = hidden2 @ w_post_ffn^T, mask, scatter-add ----------
  {
    v16h aH[16];
#pragma unroll
    for (int kt = 0; kt < 16; ++kt) aH[kt] = ldsA(&sHid[0][0], 520, kt * 32);
    const v16h* B = (const v16h*)swPostF;
#pragma unroll 1
    for (int nt = 0; nt < 16; nt += 2) {
      v8f acc0 = {}, acc1 = {};
#pragma unroll
      for (int qq = 0; qq < 4; ++qq) {
        v16h b0[4], b1[4];
#pragma unroll
        for (int q = 0; q < 4; ++q) {
          int kt = qq * 4 + q;
          b0[q] = B[(kt * 16 + nt) * 32 + lane];
          b1[q] = B[(kt * 16 + nt + 1) * 32 + lane];
        }
#pragma unroll
        for (int q = 0; q < 4; ++q) {
          int kt = qq * 4 + q;
          acc0 = WMMA16(aH[kt], b0[q], acc0);
          acc1 = WMMA16(aH[kt], b1[q], acc1);
        }
      }
      int rb = (lane >> 4) * 8;
      int n0 = nt * 16 + (lane & 15);
#pragma unroll
      for (int r = 0; r < 8; ++r) {
        int row = rb + r;
        float m = sVal[row];
        size_t base = (size_t)sK1[row] * 256;
        atomicAdd(&xx[base + n0],      acc0[r] * m);
        atomicAdd(&xx[base + n0 + 16], acc1[r] * m);
      }
    }
  }
}

// ---- finalize: out = clamp(deg,1)^deg_param * xx + x_res ------------------
__global__ void finalize_kernel(const float* __restrict__ xx,
                                const float* __restrict__ deg,
                                const float* __restrict__ dp,
                                const float* __restrict__ x_res,
                                float* __restrict__ out, long long total) {
  long long i = (long long)blockIdx.x * blockDim.x + threadIdx.x;
  long long s = (long long)gridDim.x * blockDim.x;
  for (; i < total; i += s) {
    int c   = (int)(i & 255);
    int row = (int)(i >> 8);
    float d = fmaxf(deg[row], 1.0f);
    out[i] = powf(d, dp[c]) * xx[i] + x_res[i];
  }
}

// ---------------------------------------------------------------------------
extern "C" void kernel_launch(void* const* d_in, const int* in_sizes, int n_in,
                              void* d_out, int out_size, void* d_ws, size_t ws_size,
                              hipStream_t stream) {
  const float* x        = (const float*)d_in[0];
  const float* x_res    = (const float*)d_in[1];
  const float* z_rw     = (const float*)d_in[2];
  const int*   ei       = (const int*)d_in[3];
  const float* w_src    = (const float*)d_in[4];
  const float* w_tgt    = (const float*)d_in[5];
  const float* w_pre    = (const float*)d_in[6];
  const float* w_gate_p = (const float*)d_in[7];
  const float* w_val_p  = (const float*)d_in[8];
  const float* w_post_p = (const float*)d_in[9];
  const float* w_gate_f = (const float*)d_in[10];
  const float* w_val_f  = (const float*)d_in[11];
  const float* w_post_f = (const float*)d_in[12];
  const float* deg_p    = (const float*)d_in[13];
  float* out = (float*)d_out;

  const int Ntot = in_sizes[0] / WIDTH;
  const int Etot = in_sizes[3] / 2;

  // ---- workspace layout ----
  char* ws = (char*)d_ws;
  float* x0  = (float*)ws; ws += (size_t)Ntot * 256 * 4;
  float* x1  = (float*)ws; ws += (size_t)Ntot * 256 * 4;
  float* xx  = (float*)ws; ws += (size_t)Ntot * 256 * 4;
  float* deg = (float*)ws; ws += (((size_t)Ntot * 4) + 31) & ~(size_t)31;
  _Float16* swSrc   = (_Float16*)ws; ws += (size_t)256 * 256 * 2;
  _Float16* swTgt   = (_Float16*)ws; ws += (size_t)256 * 256 * 2;
  _Float16* swPre   = (_Float16*)ws; ws += (size_t)32 * 256 * 2;
  _Float16* swPostP = (_Float16*)ws; ws += (size_t)512 * 256 * 2;
  _Float16* swPostF = (_Float16*)ws; ws += (size_t)512 * 256 * 2;
  _Float16* swGateP = (_Float16*)ws; ws += (size_t)8 * 2048 * 2;
  _Float16* swValP  = (_Float16*)ws; ws += (size_t)8 * 2048 * 2;
  _Float16* swGateF = (_Float16*)ws; ws += (size_t)8 * 2048 * 2;
  _Float16* swValF  = (_Float16*)ws; ws += (size_t)8 * 2048 * 2;

  // ---- zero accumulators (xx and deg are contiguous) ----
  {
    long long tot = (long long)Ntot * 256 + Ntot;
    int g = (int)((tot + 255) / 256); if (g > 4096) g = 4096;
    zero_f32<<<g, 256, 0, stream>>>(xx, tot);
  }

  // ---- swizzle weights into fragment-major f16 ----
  auto sw = [&](const float* w, _Float16* dst, int K, int Nout, int nh,
                int shs, int dhs) {
    dim3 grid((K * Nout + 255) / 256, nh);
    swizzle_w<<<grid, 256, 0, stream>>>(w, dst, K, Nout, shs, dhs);
  };
  sw(w_src,    swSrc,   256, 256, 1, 0, 0);
  sw(w_tgt,    swTgt,   256, 256, 1, 0, 0);
  sw(w_pre,    swPre,    32, 256, 1, 0, 0);
  sw(w_post_p, swPostP, 512, 256, 1, 0, 0);
  sw(w_post_f, swPostF, 512, 256, 1, 0, 0);
  sw(w_gate_p, swGateP,  32,  64, 8, APH * CPH, 2048);
  sw(w_val_p,  swValP,   32,  64, 8, APH * CPH, 2048);
  sw(w_gate_f, swGateF,  32,  64, 8, APH * CPH, 2048);
  sw(w_val_f,  swValF,   32,  64, 8, APH * CPH, 2048);

  // ---- node projections ----
  {
    int g = (Ntot + 15) / 16;
    gemm_node<<<g, 128, 0, stream>>>(x, swSrc, x0, Ntot);
    gemm_node<<<g, 128, 0, stream>>>(x, swTgt, x1, Ntot);
  }

  // ---- degree ----
  deg_kernel<<<(Etot + 255) / 256, 256, 0, stream>>>(ei, deg, Etot);

  // ---- fused edge pipeline ----
  edge_kernel<<<(Etot + 15) / 16, 32, 0, stream>>>(
      z_rw, ei, x0, x1, swPre, swGateP, swValP, swPostP,
      swGateF, swValF, swPostF, xx, Etot);

  // ---- finalize ----
  {
    long long tot = (long long)Ntot * 256;
    int g = (int)((tot + 255) / 256); if (g > 4096) g = 4096;
    finalize_kernel<<<g, 256, 0, stream>>>(xx, deg, deg_p, x_res, out, tot);
  }
}
